// OneDConvertedParallelMHA_77850577207451
// MI455X (gfx1250) — matrix operand order repaired and checked by reading, not compiled, not run on hardware
//
#include <hip/hip_runtime.h>

#define B_  2
#define S_  2048
#define DM_ 2048
#define H_  16
#define HD_ 128

typedef __attribute__((ext_vector_type(16))) __bf16 v16bf;
typedef __attribute__((ext_vector_type(8)))  __bf16 v8bf;
typedef __attribute__((ext_vector_type(8)))  float  v8f;

// Low 32 bits of a generic pointer to a __shared__ object == LDS byte offset
// (flat address for LDS aperture = {shared_base_hi32, lds_offset}).
#define LDS_ADDR(p) ((unsigned)(unsigned long long)(p))

// CDNA5 async global->LDS copy: 16B per active lane, tracked by ASYNCcnt.
__device__ __forceinline__ void async_b128(const void* g, unsigned lds) {
  asm volatile("global_load_async_to_lds_b128 %0, %1, off"
               :
               : "v"(lds), "v"((unsigned long long)g)
               : "memory");
}
template <int N>
__device__ __forceinline__ void wait_async() {
  asm volatile("s_wait_asynccnt %0" : : "i"(N) : "memory");
}

__device__ __forceinline__ v16bf ld_frag(const __bf16* p, int o0, int o1) {
  v8bf lo = *(const v8bf*)(p + o0);
  v8bf hi = *(const v8bf*)(p + o1);
  v16bf f;
#pragma unroll
  for (int i = 0; i < 8; ++i) { f[i] = lo[i]; f[i + 8] = hi[i]; }
  return f;
}

__device__ __forceinline__ v8f wmma_bf16(v16bf a, v16bf b, v8f c) {
  return __builtin_amdgcn_wmma_f32_16x16x32_bf16(false, a, false, b, (short)0, c,
                                                 false, false);
}

// ---------------------------------------------------------------- fp32 -> bf16
__global__ __launch_bounds__(256) void cvt_f32_bf16(const float* __restrict__ src,
                                                    __bf16* __restrict__ dst, int n) {
  int i = blockIdx.x * 1024 + threadIdx.x;
#pragma unroll
  for (int t = 0; t < 4; ++t) {
    int k = i + t * 256;
    if (k < n) dst[k] = (__bf16)src[k];
  }
}

// ------------------------------------------- C[M,N] = A[M,K] * W[N,K]^T + bias
// block tile 128x64, BK=32, 256 threads = 8 waves (4 in M, 2 in N).
// Double-buffered LDS fed by async global->LDS copies, pipelined on ASYNCcnt.
__global__ __launch_bounds__(256) void gemm_bf16_bias(
    const __bf16* __restrict__ A, const __bf16* __restrict__ W,
    const float* __restrict__ bias, float* __restrict__ C, int M, int N, int K) {
  __shared__ __align__(16) __bf16 As[2][128][32];
  __shared__ __align__(16) __bf16 Bs[2][64][32];
  const int tid  = threadIdx.x;
  const int lane = tid & 31;
  const int wave = tid >> 5;
  const int half = lane >> 4;
  const int idx  = lane & 15;
  const int wm   = wave & 3;
  const int wn   = wave >> 2;
  const int row0 = blockIdx.y * 128;
  const int col0 = blockIdx.x * 64;
  const int ao0 = 8 * half, ao1 = 16 + 8 * half;   // A-frag k offsets
  const int bo0 = 8 * half, bo1 = 16 + 8 * half;   // B-frag k offsets
  const int ra = tid >> 2, pa = tid & 3;           // tile copy coords

  // 3 async b128 copies per thread per tile (A: 2 rows halves, B: 1)
  auto issue = [&](int buf, int k0) {
    async_b128(A + (size_t)(row0 + ra) * K + k0 + pa * 8,
               LDS_ADDR(&As[buf][ra][pa * 8]));
    async_b128(A + (size_t)(row0 + ra + 64) * K + k0 + pa * 8,
               LDS_ADDR(&As[buf][ra + 64][pa * 8]));
    async_b128(W + (size_t)(col0 + ra) * K + k0 + pa * 8,
               LDS_ADDR(&Bs[buf][ra][pa * 8]));
  };

  v8f acc[2][2] = {};
  issue(0, 0);
  const int nk = K >> 5;
  for (int ki = 0; ki < nk; ++ki) {
    const int buf = ki & 1;
    __syncthreads();                 // everyone done reading buf^1
    if (ki + 1 < nk) {
      issue(buf ^ 1, (ki + 1) << 5); // prefetch next tile (overlaps compute)
      wait_async<3>();               // in-order: current tile's 3 copies landed
    } else {
      wait_async<0>();
    }
    __syncthreads();                 // current tile visible to all waves

    v16bf af[2], bfr[2];
#pragma unroll
    for (int mi = 0; mi < 2; ++mi)
      af[mi] = ld_frag(&As[buf][wm * 32 + mi * 16 + idx][0], ao0, ao1);
#pragma unroll
    for (int ni = 0; ni < 2; ++ni)
      bfr[ni] = ld_frag(&Bs[buf][wn * 32 + ni * 16 + idx][0], bo0, bo1);
#pragma unroll
    for (int mi = 0; mi < 2; ++mi)
#pragma unroll
      for (int ni = 0; ni < 2; ++ni)
        acc[mi][ni] = wmma_bf16(af[mi], bfr[ni], acc[mi][ni]);
  }

#pragma unroll
  for (int mi = 0; mi < 2; ++mi)
#pragma unroll
    for (int ni = 0; ni < 2; ++ni) {
      int col = col0 + wn * 32 + ni * 16 + idx;
      float bv = bias[col];
#pragma unroll
      for (int r = 0; r < 8; ++r) {
        int row = row0 + wm * 32 + mi * 16 + r + 8 * half;
        C[(size_t)row * N + col] = acc[mi][ni][r] + bv;
      }
    }
}

// -------------------- deinterleave + RoPE(q,k), split v; fp32 qkv -> bf16 bhsd
__global__ __launch_bounds__(64) void rope_split(const float* __restrict__ qkv,
                                                 __bf16* __restrict__ Qo,
                                                 __bf16* __restrict__ Ko,
                                                 __bf16* __restrict__ Vo) {
  int bid = blockIdx.x;
  int h = bid & (H_ - 1);
  int s = (bid >> 4) & (S_ - 1);
  int b = bid >> 15;
  int j = threadIdx.x;  // 0..63 rotary pair index

  const float* base = qkv + (size_t)(b * S_ + s) * (3 * DM_) + h * (3 * HD_);
  float q0 = base[2 * j], q1 = base[2 * j + 1];
  float k0 = base[HD_ + 2 * j], k1 = base[HD_ + 2 * j + 1];
  float v0 = base[2 * HD_ + 2 * j], v1 = base[2 * HD_ + 2 * j + 1];

  // inv_freq = 10000^(-j/64) = exp(-j*ln(10000)/64)
  float inv = __expf(-(float)j * (9.210340371976184f / 64.0f));
  float ang = (float)s * inv;
  float sn, cs;
  __sincosf(ang, &sn, &cs);

  size_t ob = ((size_t)(b * H_ + h) * S_ + s) * HD_;
  Qo[ob + j]         = (__bf16)(q0 * cs - q1 * sn);
  Qo[ob + 64 + j]    = (__bf16)(q1 * cs + q0 * sn);
  Ko[ob + j]         = (__bf16)(k0 * cs - k1 * sn);
  Ko[ob + 64 + j]    = (__bf16)(k1 * cs + k0 * sn);
  Vo[ob + 2 * j]     = (__bf16)v0;
  Vo[ob + 2 * j + 1] = (__bf16)v1;
}

// ------------------------------------------------- causal flash attention
// grid (S/64, B*H); 128 threads = 4 waves; wave w owns 16 query rows.
// Q/K tiles stream in via async global->LDS; V is transposed through VGPRs.
__global__ __launch_bounds__(128) void flash_attn(const __bf16* __restrict__ Q,
                                                  const __bf16* __restrict__ K,
                                                  const __bf16* __restrict__ V,
                                                  __bf16* __restrict__ ctx) {
  __shared__ __align__(16) __bf16 Qs[64][128];
  __shared__ __align__(16) __bf16 Ks[64][128];   // [key][d]  (B for Q*K^T)
  __shared__ __align__(16) __bf16 Vst[128][64];  // [d][key]  (B for P*V)
  __shared__ __align__(16) __bf16 Ps[4][16][64]; // per-wave P, A-layout source

  const int tid = threadIdx.x;
  const int lane = tid & 31;
  const int wave = tid >> 5;
  const int half = lane >> 4;
  const int idx = lane & 15;
  const int ao0 = 8 * half, ao1 = 16 + 8 * half;
  const int bo0 = 8 * half, bo1 = 16 + 8 * half;

  const int bh = blockIdx.y;
  const int q0 = blockIdx.x * 64;
  const __bf16* Qb = Q + (size_t)bh * S_ * HD_;
  const __bf16* Kb = K + (size_t)bh * S_ * HD_;
  const __bf16* Vb = V + (size_t)bh * S_ * HD_;

  // async load of the query tile (consumed after first wait/barrier)
#pragma unroll
  for (int it = 0; it < 8; ++it) {
    int u = tid + it * 128;
    int r = u >> 4, p = u & 15;
    async_b128(Qb + (size_t)(q0 + r) * HD_ + p * 8, LDS_ADDR(&Qs[r][p * 8]));
  }

  v8f acco[8] = {};
  float mrow[8], lrow[8];
#pragma unroll
  for (int r = 0; r < 8; ++r) { mrow[r] = -1e30f; lrow[r] = 0.f; }
  const float scale = 0.08838834764831845f;  // 1/sqrt(128)
  const int ntiles = (q0 >> 6) + 1;          // causal early exit

  for (int kt = 0; kt < ntiles; ++kt) {
    const int key0 = kt * 64;
    __syncthreads();  // previous tile's readers are done
#pragma unroll
    for (int it = 0; it < 8; ++it) {  // K tile: async straight into [key][d]
      int u = tid + it * 128;
      int r = u >> 4, p = u & 15;
      async_b128(Kb + (size_t)(key0 + r) * HD_ + p * 8,
                 LDS_ADDR(&Ks[r][p * 8]));
    }
#pragma unroll
    for (int it = 0; it < 8; ++it) {  // V tile: VGPR transpose into [d][key]
      int u = tid + it * 128;
      int r = u >> 4, p = u & 15;
      v8bf vv = *(const v8bf*)(Vb + (size_t)(key0 + r) * HD_ + p * 8);
#pragma unroll
      for (int e = 0; e < 8; ++e) Vst[p * 8 + e][r] = vv[e];
    }
    wait_async<0>();
    __syncthreads();

    // S = Q * K^T  (16x64 per wave)
    v8f sacc[4] = {};
#pragma unroll
    for (int ks = 0; ks < 4; ++ks) {
      v16bf a = ld_frag(&Qs[wave * 16 + idx][ks * 32], ao0, ao1);
#pragma unroll
      for (int j = 0; j < 4; ++j) {
        v16bf bfr = ld_frag(&Ks[j * 16 + idx][ks * 32], bo0, bo1);
        sacc[j] = wmma_bf16(a, bfr, sacc[j]);
      }
    }

    // causal mask + scale
#pragma unroll
    for (int j = 0; j < 4; ++j) {
      int kg = key0 + j * 16 + idx;
#pragma unroll
      for (int r = 0; r < 8; ++r) {
        int qg = q0 + wave * 16 + r + 8 * half;
        float v = sacc[j][r] * scale;
        sacc[j][r] = (kg <= qg) ? v : -1e30f;
      }
    }

    // online softmax (rows live in 16-lane halves -> xor-reduce with mask<=8)
#pragma unroll
    for (int r = 0; r < 8; ++r) {
      float v = fmaxf(fmaxf(sacc[0][r], sacc[1][r]), fmaxf(sacc[2][r], sacc[3][r]));
#pragma unroll
      for (int msk = 8; msk >= 1; msk >>= 1) v = fmaxf(v, __shfl_xor(v, msk, 32));
      float mnew = fmaxf(mrow[r], v);
      float corr = __expf(mrow[r] - mnew);
      mrow[r] = mnew;
      lrow[r] *= corr;
#pragma unroll
      for (int dj = 0; dj < 8; ++dj) acco[dj][r] *= corr;
    }
#pragma unroll
    for (int j = 0; j < 4; ++j)
#pragma unroll
      for (int r = 0; r < 8; ++r) sacc[j][r] = __expf(sacc[j][r] - mrow[r]);
#pragma unroll
    for (int r = 0; r < 8; ++r) {
      float v = sacc[0][r] + sacc[1][r] + sacc[2][r] + sacc[3][r];
#pragma unroll
      for (int msk = 8; msk >= 1; msk >>= 1) v += __shfl_xor(v, msk, 32);
      lrow[r] += v;
    }

    // D-layout -> A-layout via LDS
#pragma unroll
    for (int j = 0; j < 4; ++j)
#pragma unroll
      for (int r = 0; r < 8; ++r)
        Ps[wave][r + 8 * half][j * 16 + idx] = (__bf16)sacc[j][r];
    __syncthreads();

    // O += P * V
#pragma unroll
    for (int ks = 0; ks < 2; ++ks) {
      v16bf a = ld_frag(&Ps[wave][idx][ks * 32], ao0, ao1);
#pragma unroll
      for (int dj = 0; dj < 8; ++dj) {
        v16bf bfr = ld_frag(&Vst[dj * 16 + idx][ks * 32], bo0, bo1);
        acco[dj] = wmma_bf16(a, bfr, acco[dj]);
      }
    }
  }

  const int b = bh >> 4, h = bh & 15;
#pragma unroll
  for (int dj = 0; dj < 8; ++dj)
#pragma unroll
    for (int r = 0; r < 8; ++r) {
      int srow = q0 + wave * 16 + r + 8 * half;
      size_t o = ((size_t)(b * S_ + srow)) * DM_ + h * HD_ + dj * 16 + idx;
      ctx[o] = (__bf16)(acco[dj][r] / lrow[r]);
    }
}

extern "C" void kernel_launch(void* const* d_in, const int* in_sizes, int n_in,
                              void* d_out, int out_size, void* d_ws, size_t ws_size,
                              hipStream_t stream) {
  (void)in_sizes; (void)n_in; (void)out_size; (void)ws_size;
  const float* x    = (const float*)d_in[0];
  const float* Wqkv = (const float*)d_in[1];
  const float* bqkv = (const float*)d_in[2];
  const float* Wo   = (const float*)d_in[3];
  const float* bo   = (const float*)d_in[4];
  float* out = (float*)d_out;

  const int M = B_ * S_;      // 4096 token rows
  const int Nqkv = 3 * DM_;   // 6144

  char* ws = (char*)d_ws;
  auto carve = [&](size_t bytes) {
    char* p = ws;
    ws += (bytes + 255) & ~(size_t)255;
    return p;
  };
  __bf16* xb    = (__bf16*)carve((size_t)M * DM_ * 2);
  __bf16* wqkvb = (__bf16*)carve((size_t)Nqkv * DM_ * 2);
  __bf16* wob   = (__bf16*)carve((size_t)DM_ * DM_ * 2);
  float*  qkv   = (float*) carve((size_t)M * Nqkv * 4);
  __bf16* Qh    = (__bf16*)carve((size_t)M * DM_ * 2);
  __bf16* Kh    = (__bf16*)carve((size_t)M * DM_ * 2);
  __bf16* Vh    = (__bf16*)carve((size_t)M * DM_ * 2);
  __bf16* ctx   = (__bf16*)carve((size_t)M * DM_ * 2);

  {
    int n = M * DM_;
    cvt_f32_bf16<<<(n + 1023) / 1024, 256, 0, stream>>>(x, xb, n);
  }
  {
    int n = Nqkv * DM_;
    cvt_f32_bf16<<<(n + 1023) / 1024, 256, 0, stream>>>(Wqkv, wqkvb, n);
  }
  {
    int n = DM_ * DM_;
    cvt_f32_bf16<<<(n + 1023) / 1024, 256, 0, stream>>>(Wo, wob, n);
  }
  gemm_bf16_bias<<<dim3(Nqkv / 64, M / 128), 256, 0, stream>>>(xb, wqkvb, bqkv,
                                                               qkv, M, Nqkv, DM_);
  rope_split<<<B_ * S_ * H_, 64, 0, stream>>>(qkv, Qh, Kh, Vh);
  flash_attn<<<dim3(S_ / 64, B_ * H_), 128, 0, stream>>>(Qh, Kh, Vh, ctx);
  gemm_bf16_bias<<<dim3(DM_ / 64, M / 128), 256, 0, stream>>>(ctx, wob, bo, out,
                                                              M, DM_, DM_);
}
